// MultiHeadAttention_16363825398630
// MI455X (gfx1250) — matrix operand order repaired and checked by reading, not compiled, not run on hardware
//
#include <hip/hip_runtime.h>
#include <hip/hip_bf16.h>

#define B_  2
#define S_  2048
#define D_  768
#define H_  12
#define DK_ 64
#define NEG_ (-1000000000.0f)

typedef __attribute__((ext_vector_type(16))) __bf16 bf16x16;
typedef __attribute__((ext_vector_type(8)))  __bf16 bf16x8;
typedef __attribute__((ext_vector_type(4)))  __bf16 bf16x4;
typedef __attribute__((ext_vector_type(8)))  float  f32x8;
typedef __attribute__((ext_vector_type(4)))  float  f32x4;
typedef __attribute__((ext_vector_type(4)))  int    i32x4;

// ---------------------------------------------------------------------------
// Kernel 0: one-shot f32 -> bf16 conversion of Wq and Wk (2 x 768x768).
// Removes 256x-redundant per-tile weight conversion from the GEMM hot loop.
// ---------------------------------------------------------------------------
__global__ void __launch_bounds__(256)
cvtw_kernel(const float* __restrict__ Wq, const float* __restrict__ Wk,
            __bf16* __restrict__ Wqb, __bf16* __restrict__ Wkb)
{
    const int DD = D_ * D_;
    int i = (blockIdx.x * 256 + threadIdx.x) * 4;
    const float* src;
    __bf16* dst;
    int off;
    if (i < DD) { src = Wq; dst = Wqb; off = i; }
    else        { src = Wk; dst = Wkb; off = i - DD; }
    f32x4 v = *(const f32x4*)(src + off);
    bf16x4 o;
    o[0] = (__bf16)v[0]; o[1] = (__bf16)v[1];
    o[2] = (__bf16)v[2]; o[3] = (__bf16)v[3];
    *(bf16x4*)(dst + off) = o;
}

// ---------------------------------------------------------------------------
// Kernel 1: Q/K projection.  out = X @ W^T + b, stored as bf16 [B*S, D].
// One wave per 16x64 output strip: A tile converted once per K-step and
// reused by 4 WMMAs; B tiles are pre-converted bf16 (no cvt in loop).
// The 1/sqrt(DK)=0.125 score scale is folded into the Q output.
// ---------------------------------------------------------------------------
__global__ void __launch_bounds__(128)
proj_kernel(const float* __restrict__ query, const float* __restrict__ keyin,
            const __bf16* __restrict__ Wqb, const float* __restrict__ bq,
            const __bf16* __restrict__ Wkb, const float* __restrict__ bk,
            __bf16* __restrict__ Qp, __bf16* __restrict__ Kp)
{
    const bool isQ = (blockIdx.y == 0);
    const float*  X    = isQ ? query : keyin;
    const __bf16* W    = isQ ? Wqb   : Wkb;
    const float*  bias = isQ ? bq    : bk;
    __bf16* Out        = isQ ? Qp    : Kp;
    const float qs     = isQ ? 0.125f : 1.0f;        // fold 1/sqrt(DK) into Q

    const int lane = threadIdx.x & 31;
    const int wave = threadIdx.x >> 5;
    const int tile = blockIdx.x * 4 + wave;          // 256 * 12 strips total
    const int m0 = (tile / (D_ / 64)) * 16;          // row tile in [B*S)
    const int n0 = (tile % (D_ / 64)) * 64;          // col strip in [D)

    const int hl  = lane >> 4;                       // lane half (0/1)
    const int l16 = lane & 15;

    f32x8 c[4] = {{}, {}, {}, {}};
    for (int k0 = 0; k0 < D_; k0 += 32) {
        // ---- A tile 16x32 (row m0+l16 of X), contiguous k = 16*hl + e, same as B ----
        const float* arow = X + (size_t)(m0 + l16) * D_ + k0;
        f32x4 lo0 = *(const f32x4*)(arow + 16 * hl);
        f32x4 lo1 = *(const f32x4*)(arow + 16 * hl + 4);
        f32x4 hi0 = *(const f32x4*)(arow + 16 * hl + 8);
        f32x4 hi1 = *(const f32x4*)(arow + 16 * hl + 12);
        bf16x16 a;
        #pragma unroll
        for (int e = 0; e < 4; ++e) {
            a[e]      = (__bf16)lo0[e];
            a[e + 4]  = (__bf16)lo1[e];
            a[e + 8]  = (__bf16)hi0[e];
            a[e + 12] = (__bf16)hi1[e];
        }
        // ---- 4 B tiles 32x16 (bf16, direct load): col n = row n of W ----
        #pragma unroll
        for (int n = 0; n < 4; ++n) {
            const __bf16* brow =
                W + (size_t)(n0 + 16 * n + l16) * D_ + k0 + 16 * hl;
            bf16x16 bv = *(const bf16x16*)brow;
            c[n] = __builtin_amdgcn_wmma_f32_16x16x32_bf16(false, a, false, bv,
                                                           (short)0, c[n],
                                                           false, false);
        }
    }

    #pragma unroll
    for (int n = 0; n < 4; ++n) {
        const int col = n0 + 16 * n + l16;
        const float bb = bias[col];
        #pragma unroll
        for (int r = 0; r < 8; ++r) {
            const int row = m0 + r + 8 * hl;
            Out[(size_t)row * D_ + col] = (__bf16)((c[n][r] + bb) * qs);
        }
    }
}

// ---------------------------------------------------------------------------
// Kernel 2: scores + mask + syntax + softmax, one WG per (b, h, 16 q-rows).
// 16x2048 f32 score strip lives entirely in LDS (128 KB of the 320 KB WGP
// LDS); output written to HBM exactly once, 128-bit coalesced.
// ---------------------------------------------------------------------------
__global__ void __launch_bounds__(256)
attn_kernel(const __bf16* __restrict__ Qp, const __bf16* __restrict__ Kp,
            const float* __restrict__ syn, const int* __restrict__ mask,
            float* __restrict__ out)
{
    extern __shared__ float smem[];                  // [16][S_]

    const int qt = blockIdx.x % (S_ / 16);
    const int bh = blockIdx.x / (S_ / 16);
    const int h  = bh % H_;
    const int b  = bh / H_;
    const int q0 = qt * 16;

    const int lane = threadIdx.x & 31;
    const int wave = threadIdx.x >> 5;               // 8 waves
    const int hl   = lane >> 4;
    const int l16  = lane & 15;

    // ---- A operand: Q rows q0..q0+15, dk 0..63 (two 16x32 bf16 tiles) ----
    const __bf16* qrow = Qp + (size_t)(b * S_ + q0 + l16) * D_ + h * DK_;
    bf16x8 qlo0 = *(const bf16x8*)(qrow + 16 * hl);
    bf16x8 qhi0 = *(const bf16x8*)(qrow + 16 * hl + 8);
    bf16x8 qlo1 = *(const bf16x8*)(qrow + 32 + 16 * hl);
    bf16x8 qhi1 = *(const bf16x8*)(qrow + 32 + 16 * hl + 8);
    bf16x16 a0, a1;
    #pragma unroll
    for (int e = 0; e < 8; ++e) {
        a0[e] = qlo0[e]; a0[e + 8] = qhi0[e];
        a1[e] = qlo1[e]; a1[e + 8] = qhi1[e];
    }

    // ---- Phase A: QK^T into LDS.  Each wave: 16 column tiles of 16 keys ----
    for (int ct = wave; ct < S_ / 16; ct += 8) {
        const int n0 = ct * 16;
        const __bf16* krow = Kp + (size_t)(b * S_ + n0 + l16) * D_ + h * DK_;
        bf16x16 b0 = *(const bf16x16*)(krow + 16 * hl);        // dk 0..31
        bf16x16 b1 = *(const bf16x16*)(krow + 32 + 16 * hl);   // dk 32..63
        f32x8 c = {};
        c = __builtin_amdgcn_wmma_f32_16x16x32_bf16(false, a0, false, b0,
                                                    (short)0, c, false, false);
        c = __builtin_amdgcn_wmma_f32_16x16x32_bf16(false, a1, false, b1,
                                                    (short)0, c, false, false);
        #pragma unroll
        for (int r = 0; r < 8; ++r)
            smem[(r + 8 * hl) * S_ + n0 + l16] = c[r];
    }
    __syncthreads();

    // ---- Phase B: fused mask/syntax + softmax, 2 rows per wave,
    //      4 columns per lane per step (b128 LDS + global traffic).
    //      Score scale already folded into Qp. ----
    #pragma unroll
    for (int rr = 0; rr < 2; ++rr) {
        const int r = wave * 2 + rr;
        const int qrow_g = q0 + r;
        f32x4* srow = (f32x4*)(smem + r * S_);
        const i32x4* mrow = (const i32x4*)(mask + ((size_t)b * S_ + qrow_g) * S_);
        const f32x4* yrow = (const f32x4*)(syn  + ((size_t)b * S_ + qrow_g) * S_);

        float mx = -3.4e38f;
        #pragma unroll 4
        for (int i = lane; i < S_ / 4; i += 32) {
            f32x4 v  = srow[i];
            i32x4 mk = mrow[i];
            f32x4 sy = yrow[i];
            #pragma unroll
            for (int j = 0; j < 4; ++j) {
                float t = ((mk[j] == 0) ? NEG_ : v[j]) + sy[j];
                v[j] = t;
                mx = fmaxf(mx, t);
            }
            srow[i] = v;
        }
        #pragma unroll
        for (int off = 16; off > 0; off >>= 1)
            mx = fmaxf(mx, __shfl_xor(mx, off, 32));

        float sum = 0.0f;
        #pragma unroll 4
        for (int i = lane; i < S_ / 4; i += 32) {
            f32x4 v = srow[i];
            #pragma unroll
            for (int j = 0; j < 4; ++j) {
                float e = __expf(v[j] - mx);
                v[j] = e;
                sum += e;
            }
            srow[i] = v;
        }
        #pragma unroll
        for (int off = 16; off > 0; off >>= 1)
            sum += __shfl_xor(sum, off, 32);
        const float inv = 1.0f / sum;

        f32x4* orow = (f32x4*)(out + (((size_t)(b * H_ + h) * S_) + qrow_g) * S_);
        #pragma unroll 4
        for (int i = lane; i < S_ / 4; i += 32) {
            f32x4 v = srow[i];
            #pragma unroll
            for (int j = 0; j < 4; ++j) v[j] *= inv;
            orow[i] = v;
        }
    }
}

// ---------------------------------------------------------------------------
extern "C" void kernel_launch(void* const* d_in, const int* in_sizes, int n_in,
                              void* d_out, int out_size, void* d_ws, size_t ws_size,
                              hipStream_t stream)
{
    const float* query = (const float*)d_in[0];
    const float* keyin = (const float*)d_in[1];
    // d_in[2] = vm (unused)
    const float* syn   = (const float*)d_in[3];
    const int*   mask  = (const int*)d_in[4];
    const float* Wq    = (const float*)d_in[5];
    const float* bq    = (const float*)d_in[6];
    const float* Wk    = (const float*)d_in[7];
    const float* bk    = (const float*)d_in[8];

    // ws layout (bf16 elements): Qp | Kp | Wqb | Wkb  (~14.9 MB total)
    __bf16* Qp  = (__bf16*)d_ws;
    __bf16* Kp  = Qp  + (size_t)B_ * S_ * D_;
    __bf16* Wqb = Kp  + (size_t)B_ * S_ * D_;
    __bf16* Wkb = Wqb + (size_t)D_ * D_;

    // Stage 0: convert weights to bf16 once. 2*D*D/4 float4s, 256 thr/blk.
    cvtw_kernel<<<(2 * D_ * D_ / 4) / 256, 256, 0, stream>>>(Wq, Wk, Wqb, Wkb);

    // Stage 1: projections. (B*S/16)*(D/64) = 256*12 strips, 4 waves/block.
    dim3 pgrid((256 * 12) / 4, 2);
    proj_kernel<<<pgrid, 128, 0, stream>>>(query, keyin, Wqb, bq, Wkb, bk, Qp, Kp);

    // Stage 2: scores + softmax. B*H*(S/16) = 3072 blocks, 128 KB dynamic LDS.
    const int nblk = B_ * H_ * (S_ / 16);
    attn_kernel<<<nblk, 256, 16 * S_ * sizeof(float), stream>>>(Qp, Kp, syn, mask,
                                                                (float*)d_out);
}
